// TokenformerPAttention_3762391351786
// MI455X (gfx1250) — compile-verified, matrix operand order checked by reading, not run on hardware
//
#include <hip/hip_runtime.h>

typedef __attribute__((ext_vector_type(16))) __bf16 bf16x16;
typedef __attribute__((ext_vector_type(8)))  __bf16 bf16x8;
typedef __attribute__((ext_vector_type(8)))  float  f32x8;

#define E_DIM    1024
#define P_DIM    4096
#define N_ROWS   8192
#define BM       64
#define BN       128
#define BKE      64
#define NTHREADS 512

// Padded LDS row pitches (elements). Chosen so (pitch in dwords) % 64 banks is a
// small nonzero even value -> conflict-free ds_load_b128 fragment reads, while
// keeping every fragment base 16B-aligned.
#define XSP 1032   // xs rows: 2064 B = 516 dw == 4  (mod 64)
#define KSP 72     // ks rows:  144 B =  36 dw == 36 (mod 64)
#define VTP 40     // vt rows:   80 B =  20 dw == 20 (mod 64)
#define SCP 132    // sc rows:  528 B = 132 dw == 4  (mod 64)
#define ATP 136    // at rows:  272 B =  68 dw == 4  (mod 64)

#define KS_BUF_ELTS (BN * KSP)   // elements per ks ping-pong buffer

// LDS layout (bytes)
#define XS_OFF   0        // ushort [64][XSP]          132096
#define KS_OFF   132096   // ushort 2 x [128][KSP]      36864
#define VT_OFF   168960   // ushort [1024][VTP]         81920
#define SC_OFF   250880   // float  [64][SCP]           33792
#define AT_OFF   284672   // ushort [64][ATP]           17408
#define RED_OFF  302080   // float  [64][8]              2048
#define MR_OFF   304128   // float  [64]
#define LR_OFF   304384   // float  [64]
#define AL_OFF   304640   // float  [64]
#define SMEM_BYTES 304896

__device__ __forceinline__ unsigned short f2bf(float f) {
  unsigned int u = __float_as_uint(f);
  u += 0x7FFFu + ((u >> 16) & 1u);          // round-to-nearest-even
  return (unsigned short)(u >> 16);
}

// Load a 16-element bf16 A/B fragment per lane: elements [0..7] at p, [8..15] at p+16
// (CDNA5 16-bit matrix layout: VGPR0-3 = k in {h*8..h*8+7}, VGPR4-7 = {16+h*8..};
//  caller folds h*8 into p)
__device__ __forceinline__ bf16x16 ldfrag(const unsigned short* p) {
  bf16x8 lo = *(const bf16x8*)p;
  bf16x8 hi = *(const bf16x8*)(p + 16);
  return __builtin_shufflevector(lo, hi, 0,1,2,3,4,5,6,7,8,9,10,11,12,13,14,15);
}

__device__ __forceinline__ f32x8 wmma_bf16(bf16x16 a, bf16x16 b, f32x8 c) {
  return __builtin_amdgcn_wmma_f32_16x16x32_bf16(false, a, false, b, (short)0, c,
                                                 false, false);
}

__global__ void __launch_bounds__(NTHREADS, 1)
tokenformer_attn(const float* __restrict__ x, const float* __restrict__ Kw,
                 const float* __restrict__ Vw, float* __restrict__ out)
{
  extern __shared__ char smem[];
  unsigned short* xs = (unsigned short*)(smem + XS_OFF);
  unsigned short* ks = (unsigned short*)(smem + KS_OFF);
  unsigned short* vt = (unsigned short*)(smem + VT_OFF);
  float*          sc = (float*)(smem + SC_OFF);
  unsigned short* at = (unsigned short*)(smem + AT_OFF);
  float*         red = (float*)(smem + RED_OFF);
  float*        mrow = (float*)(smem + MR_OFF);
  float*        lrow = (float*)(smem + LR_OFF);
  float*        alph = (float*)(smem + AL_OFF);

  const int tid  = threadIdx.x;
  const int lane = tid & 31;
  const int wv   = tid >> 5;      // wave 0..15
  const int mg   = wv >> 2;       // M group 0..3 (16 rows each)
  const int ng   = wv & 3;        // N group 0..3
  const int l15  = lane & 15;
  const int h8   = (lane >> 4) * 8;
  const long row0 = (long)blockIdx.x * BM;

  const float scale = 0.03125f;   // 1/sqrt(1024), folded into x

  // ---- stage x tile [64][1024] as bf16 (scaled) ----
  for (int i = tid; i < BM * E_DIM / 4; i += NTHREADS) {
    const int elt = i * 4;
    const int r = elt >> 10, c = elt & (E_DIM - 1);
    const float4 v = *(const float4*)(x + (row0 + r) * (long)E_DIM + c);
    ushort4 b;
    b.x = f2bf(v.x * scale); b.y = f2bf(v.y * scale);
    b.z = f2bf(v.z * scale); b.w = f2bf(v.w * scale);
    *(ushort4*)(xs + r * XSP + c) = b;
  }
  if (tid < BM) { mrow[tid] = -1e30f; lrow[tid] = 0.0f; }

  // ---- O accumulators: 16 rows x 256 cols per wave ----
  f32x8 o[16];
#pragma unroll
  for (int i = 0; i < 16; ++i)
#pragma unroll
    for (int j = 0; j < 8; ++j) o[i][j] = 0.0f;

  __syncthreads();

  for (int pb = 0; pb < P_DIM / BN; ++pb) {
    const long p0 = (long)pb * BN;

    // ===== GEMM1: scores[64][128] = xs @ K^T, ks LDS ping-pong =====
    f32x8 c0, c1;
#pragma unroll
    for (int j = 0; j < 8; ++j) { c0[j] = 0.0f; c1[j] = 0.0f; }

    // prologue: stage e-chunk 0 into buffer 0
    __syncthreads();
    for (int i = tid; i < BN * BKE / 4; i += NTHREADS) {
      const int elt = i * 4;
      const int r = elt >> 6, c = elt & (BKE - 1);
      const float4 v = *(const float4*)(Kw + (p0 + r) * (long)E_DIM + c);
      ushort4 b;
      b.x = f2bf(v.x); b.y = f2bf(v.y); b.z = f2bf(v.z); b.w = f2bf(v.w);
      *(ushort4*)(ks + r * KSP + c) = b;
    }
    __syncthreads();

    for (int ec = 0; ec < E_DIM / BKE; ++ec) {
      const unsigned short* cur = ks + (ec & 1) * KS_BUF_ELTS;
      unsigned short*       nxt = ks + ((ec + 1) & 1) * KS_BUF_ELTS;

      // issue next chunk's global loads BEFORE compute (latency hidden by WMMAs)
      float4 pre[4];
      if (ec + 1 < E_DIM / BKE) {
#pragma unroll
        for (int j = 0; j < 4; ++j) {
          const int elt = (tid + j * NTHREADS) * 4;
          const int r = elt >> 6, c = elt & (BKE - 1);
          pre[j] = *(const float4*)(Kw + (p0 + r) * (long)E_DIM + (ec + 1) * BKE + c);
        }
      }

      // compute current chunk
#pragma unroll
      for (int kk = 0; kk < BKE / 32; ++kk) {
        const int kb = kk * 32 + h8;
        const bf16x16 a  = ldfrag(xs + (mg * 16 + l15) * XSP + ec * BKE + kb);
        const bf16x16 b0 = ldfrag(cur + (ng * 32 + l15) * KSP + kb);
        const bf16x16 b1 = ldfrag(cur + (ng * 32 + 16 + l15) * KSP + kb);
        c0 = wmma_bf16(a, b0, c0);
        c1 = wmma_bf16(a, b1, c1);
      }

      // convert + store prefetched chunk into the other buffer
      if (ec + 1 < E_DIM / BKE) {
        __syncthreads();   // all waves done reading 'nxt' (read in the previous iteration)
#pragma unroll
        for (int j = 0; j < 4; ++j) {
          const int elt = (tid + j * NTHREADS) * 4;
          const int r = elt >> 6, c = elt & (BKE - 1);
          ushort4 b;
          b.x = f2bf(pre[j].x); b.y = f2bf(pre[j].y);
          b.z = f2bf(pre[j].z); b.w = f2bf(pre[j].w);
          *(ushort4*)(nxt + r * KSP + c) = b;
        }
        __syncthreads();
      }
    }

    // write score fragments to LDS (C layout: VGPR r -> M = h8 + r, N = l15)
#pragma unroll
    for (int r = 0; r < 8; ++r) {
      sc[(mg * 16 + h8 + r) * SCP + ng * 32 + l15]      = c0[r];
      sc[(mg * 16 + h8 + r) * SCP + ng * 32 + 16 + l15] = c1[r];
    }
    __syncthreads();

    // ===== online softmax (8 threads per row) =====
    {
      const int r = tid >> 3, s = tid & 7;
      const float* srow = sc + r * SCP + s * 16;
      float pm = -1e30f;
#pragma unroll
      for (int j = 0; j < 16; ++j) pm = fmaxf(pm, srow[j]);
      red[r * 8 + s] = pm;
      __syncthreads();
      if (s == 0) {
        float bm = red[r * 8];
#pragma unroll
        for (int j = 1; j < 8; ++j) bm = fmaxf(bm, red[r * 8 + j]);
        const float om = mrow[r];
        const float nm = fmaxf(om, bm);
        mrow[r] = nm;
        alph[r] = __expf(om - nm);   // 0 on first block (om = -1e30)
      }
      __syncthreads();
      const float nm = mrow[r];
      unsigned short* arow = at + r * ATP + s * 16;
      float ps = 0.0f;
#pragma unroll
      for (int j = 0; j < 16; ++j) {
        const float e = __expf(srow[j] - nm);
        ps += e;
        arow[j] = f2bf(e);
      }
      red[r * 8 + s] = ps;
      __syncthreads();
      if (s == 0) {
        float sm = 0.0f;
#pragma unroll
        for (int j = 0; j < 8; ++j) sm += red[r * 8 + j];
        lrow[r] = lrow[r] * alph[r] + sm;
      }
    }
    __syncthreads();

    // rescale O accumulators by alpha (per M row)
    float a8[8];
#pragma unroll
    for (int r = 0; r < 8; ++r) a8[r] = alph[mg * 16 + h8 + r];
#pragma unroll
    for (int f = 0; f < 16; ++f)
#pragma unroll
      for (int r = 0; r < 8; ++r) o[f][r] *= a8[r];

    // ===== GEMM2: O += attn @ V  (K' = 128, four 32-wide steps) =====
    for (int kp = 0; kp < BN / 32; ++kp) {
      __syncthreads();  // previous vt readers done
      // stage V chunk [32 p][1024 e] -> vt[e][p] bf16.
      // Each thread: 4x4 f32 block transpose in registers, 4 x 8B stores.
      // Task map (pg fastest) + VTP pitch => 32 distinct banks per store.
      for (int i = tid; i < 2048; i += NTHREADS) {
        const int pg = i & 7;          // p sub-group of 4 rows
        const int cg = i >> 3;         // e group of 4 cols (0..255)
        const float* src = Vw + (p0 + kp * 32 + pg * 4) * (long)E_DIM + cg * 4;
        const float4 v0 = *(const float4*)(src + 0 * E_DIM);
        const float4 v1 = *(const float4*)(src + 1 * E_DIM);
        const float4 v2 = *(const float4*)(src + 2 * E_DIM);
        const float4 v3 = *(const float4*)(src + 3 * E_DIM);
        ushort4 s0, s1, s2, s3;
        s0.x = f2bf(v0.x); s0.y = f2bf(v1.x); s0.z = f2bf(v2.x); s0.w = f2bf(v3.x);
        s1.x = f2bf(v0.y); s1.y = f2bf(v1.y); s1.z = f2bf(v2.y); s1.w = f2bf(v3.y);
        s2.x = f2bf(v0.z); s2.y = f2bf(v1.z); s2.z = f2bf(v2.z); s2.w = f2bf(v3.z);
        s3.x = f2bf(v0.w); s3.y = f2bf(v1.w); s3.z = f2bf(v2.w); s3.w = f2bf(v3.w);
        *(ushort4*)(vt + (cg * 4 + 0) * VTP + pg * 4) = s0;
        *(ushort4*)(vt + (cg * 4 + 1) * VTP + pg * 4) = s1;
        *(ushort4*)(vt + (cg * 4 + 2) * VTP + pg * 4) = s2;
        *(ushort4*)(vt + (cg * 4 + 3) * VTP + pg * 4) = s3;
      }
      __syncthreads();

      // warm L2/WGP$ with next V chunk while this chunk's WMMAs run
      // (global_prefetch_b8: no VGPR cost, no counter)
      if (kp + 1 < BN / 32) {
        const float* nv = Vw + (p0 + (kp + 1) * 32) * (long)E_DIM;
#pragma unroll
        for (int j = 0; j < 4; ++j)
          __builtin_prefetch(nv + (tid + j * NTHREADS) * 16, 0, 1);
      }

      const bf16x16 a2 = ldfrag(at + (mg * 16 + l15) * ATP + kp * 32 + h8);
#pragma unroll
      for (int nt = 0; nt < 16; ++nt) {
        const int n = ng * 256 + nt * 16 + l15;
        const bf16x16 b2 = ldfrag(vt + n * VTP + h8);
        o[nt] = wmma_bf16(a2, b2, o[nt]);
      }
    }
  }

  // ===== epilogue: divide by softmax denominator and store fp32 =====
  __syncthreads();
  float inv8[8];
#pragma unroll
  for (int r = 0; r < 8; ++r) inv8[r] = 1.0f / lrow[mg * 16 + h8 + r];
#pragma unroll
  for (int nt = 0; nt < 16; ++nt) {
#pragma unroll
    for (int r = 0; r < 8; ++r) {
      const long row = row0 + mg * 16 + h8 + r;
      const int  col = ng * 256 + nt * 16 + l15;
      out[row * (long)E_DIM + col] = o[nt][r] * inv8[r];
    }
  }
}

extern "C" void kernel_launch(void* const* d_in, const int* in_sizes, int n_in,
                              void* d_out, int out_size, void* d_ws, size_t ws_size,
                              hipStream_t stream) {
  (void)in_sizes; (void)n_in; (void)out_size; (void)d_ws; (void)ws_size;
  const float* x = (const float*)d_in[0];
  const float* K = (const float*)d_in[1];
  const float* V = (const float*)d_in[2];
  float* out = (float*)d_out;

  // idempotent host-side attribute set (no stream interaction; capture-safe)
  hipFuncSetAttribute((const void*)tokenformer_attn,
                      hipFuncAttributeMaxDynamicSharedMemorySize, SMEM_BYTES);

  dim3 grid(N_ROWS / BM);   // 128 workgroups
  dim3 block(NTHREADS);     // 16 waves (wave32)
  tokenformer_attn<<<grid, block, SMEM_BYTES, stream>>>(x, K, V, out);
}
